// MultiInputsNet_67216238182882
// MI455X (gfx1250) — matrix operand order repaired and checked by reading, not compiled, not run on hardware
//
#include <hip/hip_runtime.h>
#include <hip/hip_bf16.h>
#include <stdint.h>

#define NHEADS 22
#define NCOMP  50
#define TOTBINS 28749
#define IND  2200
#define NB   2048
#define ROWS (NB*2)
#define HD1  1000
#define HD2  50
#define OUTD 13

#define KCHUNK   128                 // K halfs staged per LDS chunk
#define LDSTRIDE 136                 // halfs per LDS row (256B data + 16B bank-skew pad)

typedef __attribute__((ext_vector_type(16))) __bf16 v16bf;
typedef __attribute__((ext_vector_type(8)))  float  v8f;
typedef __attribute__((ext_vector_type(4)))  unsigned int uint4v;

// GCC-style vector matching the async-LDS builtin's parameter pointee type.
typedef int v4i_g __attribute__((__vector_size__(16)));
typedef __attribute__((address_space(1))) v4i_g* as1_v4i_p;
typedef __attribute__((address_space(3))) v4i_g* as3_v4i_p;

#if defined(__gfx1250__) && __has_builtin(__builtin_amdgcn_global_load_async_to_lds_b128)
#define HAS_ASYNC_LDS 1
#else
#define HAS_ASYNC_LDS 0
#endif

struct HeadTables {
  int loc[NHEADS + 1];
  int kp[NHEADS];
  int off[NHEADS];
  int hwp_total;
  int max_kp;
};

constexpr HeadTables make_tables() {
  HeadTables t{};
  const int bins[NHEADS] = {2490,2422,1983,1902,1815,1708,1593,1451,1384,1338,1351,
                            1333,1144,1070,1020,903,833,804,586,644,467,508};
  int acc = 0, o = 0, mx = 0;
  for (int i = 0; i < NHEADS; ++i) {
    t.loc[i] = acc; acc += bins[i];
    t.kp[i]  = ((bins[i] + 31) / 32) * 32;
    if (t.kp[i] > mx) mx = t.kp[i];
    t.off[i] = o; o += NCOMP * t.kp[i];
  }
  t.loc[NHEADS] = acc;
  t.hwp_total = o;
  t.max_kp = mx;
  return t;
}

constexpr HeadTables HT_h = make_tables();   // host-side (sizes)
__constant__ HeadTables HT = make_tables();  // device-side

__device__ __forceinline__ unsigned short f2bf(float f) {
  unsigned int u = __builtin_bit_cast(unsigned int, f);
  u += 0x7FFFu + ((u >> 16) & 1u);   // round-to-nearest-even
  return (unsigned short)(u >> 16);
}

union BFrag {
  v16bf v;
  unsigned short us[16];
  unsigned int ui[8];
  uint4v q[2];
};

__device__ __forceinline__ v8f wmma_bf16(v16bf a, v16bf b, v8f c) {
  return __builtin_amdgcn_wmma_f32_16x16x32_bf16(false, a, false, b, (short)0, c, false, false);
}

// Copy one 16B segment global -> LDS (async if the toolchain exposes it).
__device__ __forceinline__ void copy16_g2l(const unsigned short* gp, unsigned short* lp) {
#if HAS_ASYNC_LDS
  __builtin_amdgcn_global_load_async_to_lds_b128(
      (as1_v4i_p)(void*)gp, (as3_v4i_p)(void*)lp, 0, 0);
#else
  *(uint4v*)lp = *(const uint4v*)gp;
#endif
}

__device__ __forceinline__ void async_wait_all() {
#if defined(__gfx1250__) && __has_builtin(__builtin_amdgcn_s_wait_asynccnt)
  __builtin_amdgcn_s_wait_asynccnt(0);
#elif HAS_ASYNC_LDS
  asm volatile("s_wait_asynccnt 0" ::: "memory");
#endif
}

// ---------------------------------------------------------------------------
// One-time conversion kernels
// ---------------------------------------------------------------------------
__global__ void cvt_f32_bf16_kernel(const float* __restrict__ src,
                                    unsigned short* __restrict__ dst, int n) {
  int i = blockIdx.x * blockDim.x + threadIdx.x;
  if (i < n) dst[i] = f2bf(src[i]);
}

// Repack head_w into per-head (NCOMP x Kp) row-major bf16, zero-padded in K.
__global__ void repack_headw_kernel(const float* __restrict__ hw,
                                    unsigned short* __restrict__ hwp) {
  int head = blockIdx.y;
  int idx  = blockIdx.x * blockDim.x + threadIdx.x;
  int Kp   = HT.kp[head];
  int K    = HT.loc[head + 1] - HT.loc[head];
  if (idx >= NCOMP * Kp) return;
  int o = idx / Kp, k = idx - o * Kp;
  hwp[HT.off[head] + idx] =
      (k < K) ? f2bf(hw[(size_t)o * TOTBINS + HT.loc[head] + k]) : (unsigned short)0;
}

// ---------------------------------------------------------------------------
// Head stage: for head i, GEMM (4096 x K_i fp32) * (K_i x 50 bf16) -> combined
// One wave per block: 16M x 64N tile (4 WMMA accumulators), K stepped by 32.
// This kernel IS the HBM stream (471 MB fp32 input), so it reads global directly.
// ---------------------------------------------------------------------------
__global__ __launch_bounds__(32)
void head_gemm_kernel(const float* __restrict__ inp,          // (ROWS, TOTBINS) f32
                      const unsigned short* __restrict__ hwp, // packed bf16 weights
                      const float* __restrict__ head_b,       // (22,50) f32
                      unsigned short* __restrict__ comb) {    // (NB, IND) bf16
  const int lane  = threadIdx.x & 31;
  const int head  = blockIdx.y;
  const int mTile = blockIdx.x * 16;
  const int K     = HT.loc[head + 1] - HT.loc[head];
  const int Kp    = HT.kp[head];
  const unsigned short* BT = hwp + HT.off[head];  // ldb = Kp, 16B-aligned rows

  const int row  = mTile + (lane & 15);
  const int koff = (lane >> 4) * 8;
  const int nB   = lane & 15;
  const int kh   = (lane >> 4) * 8;
  const float* Arow = inp + (size_t)row * TOTBINS + HT.loc[head];

  v8f acc[4] = {};
  for (int kb = 0; kb < Kp; kb += 32) {
    BFrag fa;
    if (kb + 32 <= K) {
#pragma unroll
      for (int e = 0; e < 8; ++e) fa.us[e] = f2bf(Arow[kb + koff + e]);
#pragma unroll
      for (int e = 0; e < 8; ++e) fa.us[8 + e] = f2bf(Arow[kb + 16 + koff + e]);
    } else {
#pragma unroll
      for (int e = 0; e < 16; ++e) {
        int k = kb + ((e < 8) ? (koff + e) : (16 + koff + e - 8));
        fa.us[e] = (k < K) ? f2bf(Arow[k]) : (unsigned short)0;
      }
    }
    __builtin_prefetch((const void*)(Arow + kb + 64), 0, 1);
#pragma unroll
    for (int t = 0; t < 4; ++t) {
      int n = t * 16 + nB;
      BFrag fb;
      if (n < NCOMP) {
        const uint4v* qb = (const uint4v*)(BT + (size_t)n * Kp + kb + kh);
        fb.q[0] = qb[0];
        fb.q[1] = qb[2];
      } else {
        fb.q[0] = (uint4v){0, 0, 0, 0};
        fb.q[1] = (uint4v){0, 0, 0, 0};
      }
      acc[t] = wmma_bf16(fa.v, fb.v, acc[t]);
    }
  }

  // Epilogue: r = mTile + v + 8*(lane>>4); combined col = head*100 + (r&1)*50 + o
  const int mBase = mTile + (lane >> 4) * 8;
#pragma unroll
  for (int t = 0; t < 4; ++t) {
    int o = t * 16 + nB;
    if (o < NCOMP) {
      float bs = head_b[head * NCOMP + o];
#pragma unroll
      for (int v = 0; v < 8; ++v) {
        int r = mBase + v;
        int b = r >> 1, cch = r & 1;
        comb[(size_t)b * IND + head * (2 * NCOMP) + cch * NCOMP + o] =
            f2bf(acc[t][v] + bs);
      }
    }
  }
}

// ---------------------------------------------------------------------------
// LDS-staged bf16 WMMA GEMM: C(MxN) = A(MxK) * BT(NxK)^T + bias, optional leaky.
// Block = 4 waves. Block tile 32M x 256N; wave w owns 64 N-columns, all 32 M rows
// (2x4 accumulators). The 32x128 A tile is double-buffered in LDS via
// GLOBAL_LOAD_ASYNC_TO_LDS_B128 (ASYNCcnt) and shared by all 4 waves.
// ---------------------------------------------------------------------------
template <int ACT, int STORE_BF>
__global__ __launch_bounds__(128)
void gemm_lds_kernel(const unsigned short* __restrict__ A, int lda,
                     const unsigned short* __restrict__ BT, int ldb,
                     const float* __restrict__ bias,
                     unsigned short* __restrict__ Cbf,
                     float* __restrict__ Cf, int ldc,
                     int N, int K) {
  __shared__ __align__(16) unsigned short smem[2][32 * LDSTRIDE];

  const int tid   = threadIdx.x;
  const int lane  = tid & 31;
  const int wave  = tid >> 5;
  const int mTile = blockIdx.x * 32;
  const int n0w   = blockIdx.y * 256 + wave * 64;

  const int koff = (lane >> 4) * 8;
  const int nB   = lane & 15;
  const int kh   = (lane >> 4) * 8;

  v8f acc[2][4] = {};

  // Per-thread staging jobs: 512 x 16B segments per 32x128 tile, 4 per thread.
  auto stage = [&](int kc, unsigned short* buf) {
#pragma unroll
    for (int j = 0; j < 4; ++j) {
      int c   = tid + j * 128;
      int r   = c >> 4;        // 0..31 : tile row
      int seg = c & 15;        // 16B segment in the 256B row slice
      const unsigned short* gp = A + (size_t)(mTile + r) * lda + kc + seg * 8;
      copy16_g2l(gp, buf + r * LDSTRIDE + seg * 8);
    }
  };

  const int kchunks = K / KCHUNK;
  if (kchunks > 0) stage(0, &smem[0][0]);

  for (int ch = 0; ch < kchunks; ++ch) {
    async_wait_all();        // this wave's async copies have landed in LDS
    __syncthreads();         // publish buffer; also fences reuse of other buffer
    if (ch + 1 < kchunks) stage((ch + 1) * KCHUNK, &smem[(ch + 1) & 1][0]);

    const unsigned short* base = &smem[ch & 1][0];
#pragma unroll
    for (int ks = 0; ks < KCHUNK; ks += 32) {
      BFrag fa0, fa1;
      const unsigned short* l0 = base + (lane & 15) * LDSTRIDE + ks + koff;
      const unsigned short* l1 = base + ((lane & 15) + 16) * LDSTRIDE + ks + koff;
      fa0.q[0] = *(const uint4v*)l0;
      fa0.q[1] = *(const uint4v*)(l0 + 16);
      fa1.q[0] = *(const uint4v*)l1;
      fa1.q[1] = *(const uint4v*)(l1 + 16);
      const int kb = ch * KCHUNK + ks;
#pragma unroll
      for (int t = 0; t < 4; ++t) {
        int n = n0w + t * 16 + nB;
        BFrag fb;
        if (n < N) {
          const uint4v* qb = (const uint4v*)(BT + (size_t)n * ldb + kb + kh);
          fb.q[0] = qb[0];
          fb.q[1] = qb[2];
        } else {
          fb.q[0] = (uint4v){0, 0, 0, 0};
          fb.q[1] = (uint4v){0, 0, 0, 0};
        }
        acc[0][t] = wmma_bf16(fa0.v, fb.v, acc[0][t]);
        acc[1][t] = wmma_bf16(fa1.v, fb.v, acc[1][t]);
      }
    }
  }

  // K remainder (< KCHUNK): direct-from-global 32-step path + guarded tail.
  int kb = kchunks * KCHUNK;
  const unsigned short* Arow0 = A + (size_t)(mTile + (lane & 15)) * lda;
  const unsigned short* Arow1 = Arow0 + (size_t)16 * lda;
  for (; kb + 32 <= K; kb += 32) {
    BFrag fa0, fa1;
    fa0.q[0] = *(const uint4v*)(Arow0 + kb + koff);
    fa0.q[1] = *(const uint4v*)(Arow0 + kb + 16 + koff);
    fa1.q[0] = *(const uint4v*)(Arow1 + kb + koff);
    fa1.q[1] = *(const uint4v*)(Arow1 + kb + 16 + koff);
#pragma unroll
    for (int t = 0; t < 4; ++t) {
      int n = n0w + t * 16 + nB;
      BFrag fb;
      if (n < N) {
        const uint4v* qb = (const uint4v*)(BT + (size_t)n * ldb + kb + kh);
        fb.q[0] = qb[0];
        fb.q[1] = qb[2];
      } else {
        fb.q[0] = (uint4v){0, 0, 0, 0};
        fb.q[1] = (uint4v){0, 0, 0, 0};
      }
      acc[0][t] = wmma_bf16(fa0.v, fb.v, acc[0][t]);
      acc[1][t] = wmma_bf16(fa1.v, fb.v, acc[1][t]);
    }
  }
  if (kb < K) {
    BFrag fa0, fa1;
#pragma unroll
    for (int e = 0; e < 16; ++e) {
      int k = kb + ((e < 8) ? (koff + e) : (16 + koff + e - 8));
      fa0.us[e] = (k < K) ? Arow0[k] : (unsigned short)0;
      fa1.us[e] = (k < K) ? Arow1[k] : (unsigned short)0;
    }
#pragma unroll
    for (int t = 0; t < 4; ++t) {
      int n = n0w + t * 16 + nB;
      BFrag fb;
#pragma unroll
      for (int e = 0; e < 16; ++e) {
        int k = kb + kh + e + ((e < 8) ? 0 : 8);
        fb.us[e] = (n < N && k < K) ? BT[(size_t)n * ldb + k] : (unsigned short)0;
      }
      acc[0][t] = wmma_bf16(fa0.v, fb.v, acc[0][t]);
      acc[1][t] = wmma_bf16(fa1.v, fb.v, acc[1][t]);
    }
  }

  // Epilogue
#pragma unroll
  for (int mf = 0; mf < 2; ++mf) {
    const int mBase = mTile + mf * 16 + (lane >> 4) * 8;
#pragma unroll
    for (int t = 0; t < 4; ++t) {
      int n = n0w + t * 16 + nB;
      if (n < N) {
        float bs = bias[n];
#pragma unroll
        for (int v = 0; v < 8; ++v) {
          float val = acc[mf][t][v] + bs;
          if (ACT) val = (val > 0.f) ? val : 0.01f * val;
          int r = mBase + v;
          if (STORE_BF) Cbf[(size_t)r * ldc + n] = f2bf(val);
          else          Cf[(size_t)r * ldc + n]  = val;
        }
      }
    }
  }
}

// ---------------------------------------------------------------------------
// Final tiny layer: out(2048x13) = h2(2048x50) * w3(13x50)^T + b3 (f32)
// ---------------------------------------------------------------------------
__global__ void final_fc_kernel(const float* __restrict__ h2,
                                const float* __restrict__ w3,
                                const float* __restrict__ b3,
                                float* __restrict__ out) {
  int i = blockIdx.x * blockDim.x + threadIdx.x;
  if (i >= NB * OUTD) return;
  int b = i / OUTD, j = i - b * OUTD;
  float s = b3[j];
#pragma unroll
  for (int k = 0; k < HD2; ++k) s += h2[b * HD2 + k] * w3[j * HD2 + k];
  out[i] = s;
}

// ---------------------------------------------------------------------------
extern "C" void kernel_launch(void* const* d_in, const int* in_sizes, int n_in,
                              void* d_out, int out_size, void* d_ws, size_t ws_size,
                              hipStream_t stream) {
  const float* input  = (const float*)d_in[0];
  const float* head_w = (const float*)d_in[1];
  const float* head_b = (const float*)d_in[2];
  const float* conc_w = (const float*)d_in[3];
  const float* conc_b = (const float*)d_in[4];
  const float* w1     = (const float*)d_in[5];
  const float* b1     = (const float*)d_in[6];
  const float* w2     = (const float*)d_in[7];
  const float* b2     = (const float*)d_in[8];
  const float* w3     = (const float*)d_in[9];
  const float* b3     = (const float*)d_in[10];
  float* out = (float*)d_out;

  char* p = (char*)d_ws;
  auto carve = [&](size_t bytes) -> char* {
    char* r = p;
    p += (bytes + 255) & ~(size_t)255;
    return r;
  };
  unsigned short* cwb  = (unsigned short*)carve((size_t)IND * IND * 2);
  unsigned short* w1b  = (unsigned short*)carve((size_t)HD1 * IND * 2);
  unsigned short* w2b  = (unsigned short*)carve((size_t)HD2 * HD1 * 2);
  unsigned short* hwp  = (unsigned short*)carve((size_t)HT_h.hwp_total * 2);
  unsigned short* comb = (unsigned short*)carve((size_t)NB * IND * 2);
  unsigned short* zb   = (unsigned short*)carve((size_t)NB * IND * 2);
  unsigned short* h1b  = (unsigned short*)carve((size_t)NB * HD1 * 2);
  float*          h2f  = (float*)carve((size_t)NB * HD2 * 4);

  // Weight conversion / repack (small; once per launch, deterministic)
  {
    int n = IND * IND;
    cvt_f32_bf16_kernel<<<(n + 255) / 256, 256, 0, stream>>>(conc_w, cwb, n);
  }
  {
    int n = HD1 * IND;
    cvt_f32_bf16_kernel<<<(n + 255) / 256, 256, 0, stream>>>(w1, w1b, n);
  }
  {
    int n = HD2 * HD1;
    cvt_f32_bf16_kernel<<<(n + 255) / 256, 256, 0, stream>>>(w2, w2b, n);
  }
  {
    dim3 g((NCOMP * HT_h.max_kp + 255) / 256, NHEADS);
    repack_headw_kernel<<<g, 256, 0, stream>>>(head_w, hwp);
  }

  // Stage 1: per-head ragged GEMMs -> combined (bf16)
  {
    dim3 g(ROWS / 16, NHEADS);
    head_gemm_kernel<<<g, 32, 0, stream>>>(input, hwp, head_b, comb);
  }
  // Stage 2: z = combined @ conc_w^T + conc_b   (no activation)
  {
    dim3 g(NB / 32, (IND + 255) / 256);
    gemm_lds_kernel<0, 1><<<g, 128, 0, stream>>>(comb, IND, cwb, IND, conc_b,
                                                 zb, nullptr, IND, IND, IND);
  }
  // Stage 3: h1 = leaky(z @ w1^T + b1)
  {
    dim3 g(NB / 32, (HD1 + 255) / 256);
    gemm_lds_kernel<1, 1><<<g, 128, 0, stream>>>(zb, IND, w1b, IND, b1,
                                                 h1b, nullptr, HD1, HD1, IND);
  }
  // Stage 4: h2 = leaky(h1 @ w2^T + b2)  -> f32
  {
    dim3 g(NB / 32, 1);
    gemm_lds_kernel<1, 0><<<g, 128, 0, stream>>>(h1b, HD1, w2b, HD1, b2,
                                                 nullptr, h2f, HD2, HD2, HD1);
  }
  // Stage 5: out = h2 @ w3^T + b3  (f32 scalar)
  {
    int n = NB * OUTD;
    final_fc_kernel<<<(n + 255) / 256, 256, 0, stream>>>(h2f, w3, b3, out);
  }
}
